// Attention_40003325395042
// MI455X (gfx1250) — matrix operand order repaired and checked by reading, not compiled, not run on hardware
//
// CDNA5 / gfx1250 fused 2D-rel-bias multi-head attention.
// Pipeline: prep (weights transpose + query f32->f16) -> qkv_proj (WMMA GEMM, 16x64/wave)
//        -> attention (flash-style online softmax, WMMA QK^T & PV)
//        -> out_proj (WMMA GEMM + bias, 16x64/wave).
// All matrix math uses v_wmma_f32_16x16x32_f16 (f32 accumulate), softmax in f32 VALU.
#include <hip/hip_runtime.h>
#include <hip/hip_bf16.h>

typedef __attribute__((ext_vector_type(16))) _Float16 v16h;
typedef __attribute__((ext_vector_type(8)))  _Float16 h8;
typedef __attribute__((ext_vector_type(8)))  float    v8f;

#define WMMA_F16(A, B, C) \
  __builtin_amdgcn_wmma_f32_16x16x32_f16(false, (A), false, (B), (short)0, (C), false, false)

// ---- problem constants -----------------------------------------------------
constexpr int BATCH = 8;
constexpr int SEQ   = 1024;   // S = 32*32
constexpr int DMODEL= 512;
constexpr int NH    = 16;     // heads
constexpr int HK    = 32;     // head dim
constexpr int ROWS  = BATCH * SEQ;          // 8192
constexpr float QSCALE = 0.17677669529663689f; // 32^-0.5

// ---- workspace layout (bytes) ---------------------------------------------
constexpr size_t OFF_WT   = 0;                               // wt_q|wt_k|wt_v : 3*512*512 f16
constexpr size_t OFF_WOT  = 3u * 512 * 512 * 2;              // 1572864
constexpr size_t OFF_QH   = OFF_WOT + 512u * 512 * 2;        // 2097152
constexpr size_t OFF_KH   = OFF_QH  + (size_t)ROWS * 512 * 2;
constexpr size_t OFF_VT   = OFF_KH  + (size_t)ROWS * 512 * 2;
constexpr size_t OFF_AO   = OFF_VT  + (size_t)ROWS * 512 * 2;
constexpr size_t OFF_QHLF = OFF_AO  + (size_t)ROWS * 512 * 2; // query as f16 row-major

// ---- fragment loaders ------------------------------------------------------
// A-matrix 16-bit 16x32: lanes 0-15 row=lane,
// halves = K{koff..koff+15} contiguous; koff = 0 / 16 per half-wave.
__device__ __forceinline__ v16h load_a_h16(const _Float16* p) {
  return *reinterpret_cast<const v16h*>(p);
}
// B-matrix 16-bit 32x16, same contiguous k order as A (this code's choice): lane<16
// col=lane K=0..15 contiguous; lane>=16 col=lane-16 K=16..31 -> one 32B load.
__device__ __forceinline__ v16h load_b_h16(const _Float16* p) {
  return *reinterpret_cast<const v16h*>(p);
}

// ---- kernel 0: prep — weight transpose + query f16 convert ------------------
// Wq/Wk/Wv [D, N*K] -> wt[w][nk*512 + d];  Wo [N*K, D] -> wot[d*512 + nk];
// query f32 [8192*512] -> qhalf f16 (row-major, contiguous copy).
__global__ __launch_bounds__(256) void prep(
    const float* __restrict__ Wq, const float* __restrict__ Wk,
    const float* __restrict__ Wv, const float* __restrict__ Wo,
    const float* __restrict__ query,
    _Float16* __restrict__ wt_all, _Float16* __restrict__ wot,
    _Float16* __restrict__ qhalf) {
  int idx = blockIdx.x * 256 + threadIdx.x;   // 4*262144 + 8192*512 total
  if (idx < 1048576) {
    int mat = idx >> 18;
    int e   = idx & 262143;
    int r   = e >> 9;        // source row
    int c   = e & 511;       // source col
    if (mat == 0)      wt_all[           c * 512 + r] = (_Float16)Wq[r * 512 + c];
    else if (mat == 1) wt_all[262144 +   c * 512 + r] = (_Float16)Wk[r * 512 + c];
    else if (mat == 2) wt_all[524288 +   c * 512 + r] = (_Float16)Wv[r * 512 + c];
    else               wot[              c * 512 + r] = (_Float16)Wo[r * 512 + c];
  } else {
    size_t e = (size_t)idx - 1048576;          // [0, 4194304)
    qhalf[e] = (_Float16)query[e];
  }
}

// ---- kernel 1: fused QKV projection ---------------------------------------
// One wave computes a 16x64 tile (4 accumulators, A reused 4x) of one of the
// 3 GEMMs [8192x512]x[512x512].
__global__ __launch_bounds__(128) void qkv_proj(
    const _Float16* __restrict__ qhalf, const _Float16* __restrict__ wt_all,
    const float* __restrict__ bq, const float* __restrict__ bk, const float* __restrict__ bv,
    _Float16* __restrict__ qh, _Float16* __restrict__ kh, _Float16* __restrict__ vt) {
  int wave = threadIdx.x >> 5, lane = threadIdx.x & 31;
  int gw  = blockIdx.x * 4 + wave;       // [0, 3*512*8)
  int w   = gw >> 12;                    // which weight (0=q,1=k,2=v)
  int rem = gw & 4095;
  int m0  = (rem >> 3) << 4;             // row tile base (16 rows)
  int n0  = (rem & 7) << 6;              // col group base (64 cols)
  int l16  = lane & 15;
  int koff = (lane < 16) ? 0 : 16;       // A-fragment half
  int boff = (lane < 16) ? 0 : 16;       // B-fragment half
  int arow = m0 + l16;
  int col0 = n0 + l16;

  const _Float16* wmat = wt_all + (size_t)w * 262144;
  const _Float16* ap_base = qhalf + (size_t)arow * DMODEL;
  v8f acc0 = {0.f,0.f,0.f,0.f,0.f,0.f,0.f,0.f};
  v8f acc1 = acc0, acc2 = acc0, acc3 = acc0;
#pragma unroll 4
  for (int kk = 0; kk < DMODEL; kk += 32) {
    v16h a  = load_a_h16(ap_base + kk + koff);
    v16h b0 = load_b_h16(wmat + (size_t)(col0     ) * DMODEL + kk + boff);
    v16h b1 = load_b_h16(wmat + (size_t)(col0 + 16) * DMODEL + kk + boff);
    v16h b2 = load_b_h16(wmat + (size_t)(col0 + 32) * DMODEL + kk + boff);
    v16h b3 = load_b_h16(wmat + (size_t)(col0 + 48) * DMODEL + kk + boff);
    acc0 = WMMA_F16(a, b0, acc0);
    acc1 = WMMA_F16(a, b1, acc1);
    acc2 = WMMA_F16(a, b2, acc2);
    acc3 = WMMA_F16(a, b3, acc3);
  }

  const float* bias = (w == 0) ? bq : (w == 1) ? bk : bv;
  float scale = (w == 0) ? QSCALE : 1.0f;
  int half8 = (lane >> 4) << 3;
  v8f accs[4] = {acc0, acc1, acc2, acc3};
#pragma unroll
  for (int t = 0; t < 4; ++t) {
    int col   = col0 + 16 * t;
    float bcol = bias[col];
    int n = col >> 5, khead = col & 31;
#pragma unroll
    for (int j = 0; j < 8; ++j) {
      int r = m0 + j + half8;            // C layout: VGPR j -> row j (+8 for hi half)
      int bb = r >> 10, s = r & 1023;
      _Float16 hv = (_Float16)((accs[t][j] + bcol) * scale);
      size_t bn = (size_t)bb * NH + n;
      if (w == 0)      qh[(bn * SEQ + s) * HK + khead] = hv;
      else if (w == 1) kh[(bn * SEQ + s) * HK + khead] = hv;
      else             vt[(bn * HK + khead) * SEQ + s] = hv;  // transposed for PV B-frags
    }
  }
}

// ---- kernel 2: flash attention with 2D relative bias ----------------------
// One wave per (batch, head, 16-query-row block); 32-key chunks, online softmax.
__global__ __launch_bounds__(256) void attention(
    const _Float16* __restrict__ qh, const _Float16* __restrict__ kh,
    const _Float16* __restrict__ vt, const float* __restrict__ rel_bias,
    _Float16* __restrict__ ao) {
  __shared__ __align__(32) _Float16 pstage[8 * 16 * 32];   // 1KB per wave
  int wave = threadIdx.x >> 5, lane = threadIdx.x & 31;
  int gw  = blockIdx.x * 8 + wave;       // [0, 8*16*64)
  int qb  = gw & 63;
  int n   = (gw >> 6) & 15;
  int b   = gw >> 10;
  size_t bn = (size_t)b * NH + n;
  int qs0 = qb << 4;
  int l16  = lane & 15;
  int koff = (lane < 16) ? 0 : 16;
  int boff = (lane < 16) ? 0 : 16;
  int half8 = (lane >> 4) << 3;

  // Q A-fragment: row = l16, head-dims koff..koff+15 contiguous.
  v16h qa = load_a_h16(qh + (bn * SEQ + qs0 + l16) * HK + koff);

  float m[8], lsum[8];
  v8f o0 = {0.f,0.f,0.f,0.f,0.f,0.f,0.f,0.f};
  v8f o1 = {0.f,0.f,0.f,0.f,0.f,0.f,0.f,0.f};
#pragma unroll
  for (int j = 0; j < 8; ++j) { m[j] = -3.0e38f; lsum[j] = 0.f; }

  const float* rb = rel_bias + n * 3969;  // [63,63] table for this head
  _Float16* ps = pstage + wave * 512;

  for (int t0 = 0; t0 < SEQ; t0 += 32) {
    // K^T B-fragments: col = key, K-dim = head dim (contiguous 16 halves).
    v16h kb0 = load_b_h16(kh + (bn * SEQ + t0      + l16) * HK + boff);
    v16h kb1 = load_b_h16(kh + (bn * SEQ + t0 + 16 + l16) * HK + boff);
    v8f z = {0.f,0.f,0.f,0.f,0.f,0.f,0.f,0.f};
    v8f s0 = WMMA_F16(qa, kb0, z);
    v8f s1 = WMMA_F16(qa, kb1, z);

    // relative position bias gather (63x63 table is L1/L2 resident)
    int tA = t0 + l16, tB = tA + 16;
    int htA = tA >> 5, wtA = tA & 31, htB = tB >> 5, wtB = tB & 31;
#pragma unroll
    for (int j = 0; j < 8; ++j) {
      int sr = qs0 + half8 + j;
      int hs = sr >> 5, wsv = sr & 31;
      s0[j] += rb[(hs - htA + 31) * 63 + (wsv - wtA + 31)];
      s1[j] += rb[(hs - htB + 31) * 63 + (wsv - wtB + 31)];
    }

    // online softmax; each half-wave owns disjoint rows, xor 1/2/4/8 stays in-half
#pragma unroll
    for (int j = 0; j < 8; ++j) {
      float rm = fmaxf(s0[j], s1[j]);
      rm = fmaxf(rm, __shfl_xor(rm, 1, 32));
      rm = fmaxf(rm, __shfl_xor(rm, 2, 32));
      rm = fmaxf(rm, __shfl_xor(rm, 4, 32));
      rm = fmaxf(rm, __shfl_xor(rm, 8, 32));
      float mn = fmaxf(m[j], rm);
      float alpha = __expf(m[j] - mn);
      m[j] = mn;
      float p0 = __expf(s0[j] - mn);
      float p1 = __expf(s1[j] - mn);
      s0[j] = p0; s1[j] = p1;
      float rs = p0 + p1;
      rs += __shfl_xor(rs, 1, 32);
      rs += __shfl_xor(rs, 2, 32);
      rs += __shfl_xor(rs, 4, 32);
      rs += __shfl_xor(rs, 8, 32);
      lsum[j] = lsum[j] * alpha + rs;
      o0[j] *= alpha;
      o1[j] *= alpha;
    }

    // C-layout -> A-layout transpose of P through per-wave LDS slice
#pragma unroll
    for (int j = 0; j < 8; ++j) {
      int r = j + half8;
      ps[r * 32 +      l16] = (_Float16)s0[j];
      ps[r * 32 + 16 + l16] = (_Float16)s1[j];
    }
    __builtin_amdgcn_wave_barrier();   // same-wave DS ops are in-order; fence the scheduler
    v16h pa = load_a_h16(ps + l16 * 32 + koff);

    // V B-fragments from transposed layout: col = head dim, K = key (contiguous)
    v16h vb0 = load_b_h16(vt + (bn * HK +      l16) * SEQ + t0 + boff);
    v16h vb1 = load_b_h16(vt + (bn * HK + 16 + l16) * SEQ + t0 + boff);
    o0 = WMMA_F16(pa, vb0, o0);
    o1 = WMMA_F16(pa, vb1, o1);
  }

  // finalize: divide by l, store attn_out as f16 [B,S,N*K]
#pragma unroll
  for (int j = 0; j < 8; ++j) {
    float inv = 1.0f / lsum[j];
    int sr = qs0 + half8 + j;
    size_t orow = ((size_t)b * SEQ + sr) * (NH * HK) + n * HK;
    ao[orow +      l16] = (_Float16)(o0[j] * inv);
    ao[orow + 16 + l16] = (_Float16)(o1[j] * inv);
  }
}

// ---- kernel 3: output projection [8192x512]x[512x512] + bias --------------
// One wave computes a 16x64 tile (4 accumulators, A reused 4x).
__global__ __launch_bounds__(128) void out_proj(
    const _Float16* __restrict__ ao, const _Float16* __restrict__ wot,
    const float* __restrict__ bo, float* __restrict__ out) {
  int wave = threadIdx.x >> 5, lane = threadIdx.x & 31;
  int gw = blockIdx.x * 4 + wave;        // [0, 512*8)
  int m0 = (gw >> 3) << 4;
  int n0 = (gw & 7) << 6;
  int l16  = lane & 15;
  int koff = (lane < 16) ? 0 : 16;
  int boff = (lane < 16) ? 0 : 16;
  int arow = m0 + l16;
  int col0 = n0 + l16;

  const _Float16* ap_base = ao + (size_t)arow * DMODEL;
  v8f acc0 = {0.f,0.f,0.f,0.f,0.f,0.f,0.f,0.f};
  v8f acc1 = acc0, acc2 = acc0, acc3 = acc0;
#pragma unroll 4
  for (int kk = 0; kk < DMODEL; kk += 32) {
    v16h a  = load_a_h16(ap_base + kk + koff);
    v16h b0 = load_b_h16(wot + (size_t)(col0     ) * DMODEL + kk + boff);
    v16h b1 = load_b_h16(wot + (size_t)(col0 + 16) * DMODEL + kk + boff);
    v16h b2 = load_b_h16(wot + (size_t)(col0 + 32) * DMODEL + kk + boff);
    v16h b3 = load_b_h16(wot + (size_t)(col0 + 48) * DMODEL + kk + boff);
    acc0 = WMMA_F16(a, b0, acc0);
    acc1 = WMMA_F16(a, b1, acc1);
    acc2 = WMMA_F16(a, b2, acc2);
    acc3 = WMMA_F16(a, b3, acc3);
  }
  int half8 = (lane >> 4) << 3;
  v8f accs[4] = {acc0, acc1, acc2, acc3};
#pragma unroll
  for (int t = 0; t < 4; ++t) {
    int col = col0 + 16 * t;
    float bb = bo[col];
#pragma unroll
    for (int j = 0; j < 8; ++j) {
      int r = m0 + j + half8;
      out[(size_t)r * DMODEL + col] = accs[t][j] + bb;
    }
  }
}

// ---- host launcher ---------------------------------------------------------
extern "C" void kernel_launch(void* const* d_in, const int* in_sizes, int n_in,
                              void* d_out, int out_size, void* d_ws, size_t ws_size,
                              hipStream_t stream) {
  const float* query = (const float*)d_in[0];
  const float* Wq    = (const float*)d_in[1];
  const float* bq    = (const float*)d_in[2];
  const float* Wk    = (const float*)d_in[3];
  const float* bk    = (const float*)d_in[4];
  const float* Wv    = (const float*)d_in[5];
  const float* bv    = (const float*)d_in[6];
  const float* Wo    = (const float*)d_in[7];
  const float* bo    = (const float*)d_in[8];
  const float* rel   = (const float*)d_in[9];

  char* ws = (char*)d_ws;
  _Float16* wt_all = (_Float16*)(ws + OFF_WT);
  _Float16* wot    = (_Float16*)(ws + OFF_WOT);
  _Float16* qh     = (_Float16*)(ws + OFF_QH);
  _Float16* kh     = (_Float16*)(ws + OFF_KH);
  _Float16* vt     = (_Float16*)(ws + OFF_VT);
  _Float16* ao     = (_Float16*)(ws + OFF_AO);
  _Float16* qhalf  = (_Float16*)(ws + OFF_QHLF);

  // prep: 4*262144 weight elems + 8192*512 query elems = 5242880 threads
  prep<<<20480, 256, 0, stream>>>(Wq, Wk, Wv, Wo, query, wt_all, wot, qhalf);
  // 3 weights * 512 M-tiles * 8 N-groups = 12288 waves, 4 waves/block
  qkv_proj<<<3072, 128, 0, stream>>>(qhalf, wt_all, bq, bk, bv, qh, kh, vt);
  // 8*16*64 = 8192 waves, 8 waves/block
  attention<<<1024, 256, 0, stream>>>(qh, kh, vt, rel, ao);
  // 512 M-tiles * 8 N-groups = 4096 waves, 4 waves/block
  out_proj<<<1024, 128, 0, stream>>>(ao, wot, bo, (float*)d_out);
}
